// FunctionDependencyGNN_67912022884495
// MI455X (gfx1250) — matrix operand order repaired and checked by reading, not compiled, not run on hardware
//
#include <hip/hip_runtime.h>
#include <math.h>

#define D 256
#define NGRAPH 64

typedef __attribute__((ext_vector_type(16))) __bf16 v16bf;
typedef __attribute__((ext_vector_type(8)))  float  v8f;

union FragBF { v16bf v; unsigned int u[8]; };

__device__ __forceinline__ unsigned int pack_bf16x2(float lo, float hi) {
  unsigned int a = __float_as_uint(lo);
  unsigned int b = __float_as_uint(hi);
  a = (a + 0x7FFFu + ((a >> 16) & 1u)) >> 16;   // RTNE f32 -> bf16
  b = (b + 0x7FFFu + ((b >> 16) & 1u)) >> 16;
  return (b << 16) | (a & 0xFFFFu);
}

// order-preserving float <-> uint encoding (for atomic max via u32)
__device__ __forceinline__ unsigned int fenc(float f) {
  unsigned int u = __float_as_uint(f);
  return (u & 0x80000000u) ? ~u : (u | 0x80000000u);
}
__device__ __forceinline__ float fdec(unsigned int e) {
  unsigned int u = (e & 0x80000000u) ? (e & 0x7FFFFFFFu) : ~e;
  return __uint_as_float(u);
}
__device__ __forceinline__ float lrelu(float a) { return a > 0.f ? a : 0.2f * a; }

// ---------------------------------------------------------------------------
// Weight pre-pack: W[K,N] f32 row-major -> bf16 WMMA B-fragments, fragment-major
// layout Bpk[tn][kc][lane][v] (uint = 2 bf16), tn = n/16, kc = k/32.
// B-fragment layout per ISA 7.12.2: lane -> n = lane%16, koff = 8*(lane/16);
// VGPR v holds K = kc*32 + 16*(v/4) + koff + 2*(v%4) (+1 in high half).
// ---------------------------------------------------------------------------
__global__ void pack_B_k(const float* __restrict__ W, unsigned int* __restrict__ Bpk,
                         int N, int K) {
  int idx = blockIdx.x * blockDim.x + threadIdx.x;
  int KC = K >> 5;
  int total = (N >> 4) * KC * 256;
  if (idx >= total) return;
  int v    = idx & 7;
  int lane = (idx >> 3) & 31;
  int rest = idx >> 8;
  int kc   = rest % KC;
  int tn   = rest / KC;
  int n    = (tn << 4) + (lane & 15);
  int kb   = (kc << 5) + ((v >> 2) << 4) + ((lane >> 4) << 3) + ((v & 3) << 1);
  Bpk[idx] = pack_bf16x2(W[(size_t)kb * N + n], W[(size_t)(kb + 1) * N + n]);
}

// ---------------------------------------------------------------------------
// GEMM: C[M,N] = act( (A[M,K] (+Abias)) @ W[K,N] + bias ), bf16 WMMA, f32 acc.
// One wave computes a 16x64 output strip (4 column tiles): the A fragment is
// packed once per K-chunk and reused by 4 WMMAs. B comes pre-packed (b128 loads).
// M % 16 == 0, N % 64 == 0, K % 32 == 0.
// ---------------------------------------------------------------------------
template <int RELU, int HAS_ABIAS>
__global__ void gemm_wmma_bf16(const float* __restrict__ A,
                               const float* __restrict__ Abias,
                               const unsigned int* __restrict__ Bpk,
                               const float* __restrict__ bias,
                               float* __restrict__ C,
                               int M, int N, int K) {
  const int lane     = threadIdx.x & 31;
  const int wave     = threadIdx.x >> 5;
  const int tilesN64 = N >> 6;
  const int task     = blockIdx.x * (blockDim.x >> 5) + wave;
  const int nTasks   = (M >> 4) * tilesN64;
  if (task >= nTasks) return;                 // wave-uniform: EXEC stays full

  const int tm  = task / tilesN64;
  const int tn0 = (task % tilesN64) << 2;     // first of 4 column tiles

  const int mrow   = (tm << 4) + (lane & 15); // A row for this lane
  const int khalfA = (lane >> 4) << 3;        // A: lanes 16-31 hold K+8 pairs
  const int ncol0  = lane & 15;
  const int KC     = K >> 5;

  const float* __restrict__ arow = A + (size_t)mrow * K;

  v8f acc[4] = {{}, {}, {}, {}};
  for (int kc = 0; kc < KC; ++kc) {
    const int k0 = kc << 5;
    FragBF fa;
#pragma unroll
    for (int g = 0; g < 2; ++g) {
      const int ka = k0 + (g << 4) + khalfA;  // 32B-aligned, 8 contiguous floats
      float4 a0 = *(const float4*)(arow + ka);
      float4 a1 = *(const float4*)(arow + ka + 4);
      if (HAS_ABIAS) {
        float4 b0 = *(const float4*)(Abias + ka);
        float4 b1 = *(const float4*)(Abias + ka + 4);
        a0.x += b0.x; a0.y += b0.y; a0.z += b0.z; a0.w += b0.w;
        a1.x += b1.x; a1.y += b1.y; a1.z += b1.z; a1.w += b1.w;
      }
      fa.u[(g << 2) + 0] = pack_bf16x2(a0.x, a0.y);
      fa.u[(g << 2) + 1] = pack_bf16x2(a0.z, a0.w);
      fa.u[(g << 2) + 2] = pack_bf16x2(a1.x, a1.y);
      fa.u[(g << 2) + 3] = pack_bf16x2(a1.z, a1.w);
    }
#pragma unroll
    for (int t = 0; t < 4; ++t) {
      FragBF fb;
      const unsigned int* bp =
          Bpk + ((((size_t)(tn0 + t) * KC + kc) << 5) + lane) * 8;
      *(uint4*)&fb.u[0] = *(const uint4*)bp;
      *(uint4*)&fb.u[4] = *(const uint4*)(bp + 4);
      acc[t] = __builtin_amdgcn_wmma_f32_16x16x32_bf16(
          false, fa.v, false, fb.v, (short)0, acc[t], false, false);
    }
  }

  const int rbase = (tm << 4) + ((lane >> 4) << 3);  // D: row = rbase + i
#pragma unroll
  for (int t = 0; t < 4; ++t) {
    const int ncol = ((tn0 + t) << 4) + ncol0;
    const float bb = bias ? bias[ncol] : 0.f;
#pragma unroll
    for (int i = 0; i < 8; ++i) {
      float v = acc[t][i] + bb;
      if (RELU) v = fmaxf(v, 0.f);
      C[(size_t)(rbase + i) * N + ncol] = v;
    }
  }
}

// ---------------------------------------------------------------------------
// Graph kernels
// ---------------------------------------------------------------------------
__global__ void deg_init_k(float* deg, int N) {
  int i = blockIdx.x * blockDim.x + threadIdx.x;
  if (i < N) deg[i] = 1.0f;  // self-loop weight
}

// edge MLP: ew = sigmoid(relu(emb @ W1 + b1) @ W2 + b2); deg[dst] += ew
__global__ void edge_weight_k(const int* __restrict__ attr, const int* __restrict__ ei,
                              const float* __restrict__ emb,
                              const float* __restrict__ W1, const float* __restrict__ b1,
                              const float* __restrict__ W2, const float* __restrict__ b2,
                              float* __restrict__ ew, float* __restrict__ deg, int E) {
  int e = blockIdx.x * blockDim.x + threadIdx.x;
  if (e >= E) return;
  const float* t = emb + attr[e] * 64;
  float z = b2[0];
  for (int j = 0; j < 32; ++j) {
    float h = b1[j];
    for (int i = 0; i < 64; ++i) h += t[i] * W1[i * 32 + j];
    z += fmaxf(h, 0.f) * W2[j];
  }
  float w = 1.f / (1.f + expf(-z));
  ew[e] = w;
  atomicAdd(&deg[ei[E + e]], w);
}

__global__ void dis_k(float* deg, int N) {  // in-place deg -> D^{-1/2}
  int i = blockIdx.x * blockDim.x + threadIdx.x;
  if (i < N) deg[i] = rsqrtf(deg[i]);
}

__global__ void norm_k(const int* __restrict__ ei, const float* __restrict__ ew,
                       const float* __restrict__ dis, float* __restrict__ nrm, int E) {
  int e = blockIdx.x * blockDim.x + threadIdx.x;
  if (e >= E) return;
  nrm[e] = dis[ei[e]] * ew[e] * dis[ei[E + e]];
}

// a_s[n,h], a_d[n,h]
__global__ void att_logits_k(const float* __restrict__ xh, const float* __restrict__ as_w,
                             const float* __restrict__ ad_w, float* __restrict__ as_,
                             float* __restrict__ ad_, int N) {
  int idx = blockIdx.x * blockDim.x + threadIdx.x;
  if (idx >= N * 4) return;
  int n = idx >> 2, h = idx & 3;
  const float* row = xh + (size_t)n * D + h * 64;
  float s = 0.f, d = 0.f;
  for (int c = 0; c < 64; ++c) { s += row[c] * as_w[h * 64 + c]; d += row[c] * ad_w[h * 64 + c]; }
  as_[idx] = s; ad_[idx] = d;
}

__global__ void att_minit_k(const float* __restrict__ as_, const float* __restrict__ ad_,
                            unsigned int* __restrict__ menc, int N) {
  int idx = blockIdx.x * blockDim.x + threadIdx.x;
  if (idx >= N * 4) return;
  menc[idx] = fenc(lrelu(as_[idx] + ad_[idx]));  // self loop seeds the max
}

__global__ void att_edge_max_k(const int* __restrict__ ei, const float* __restrict__ as_,
                               const float* __restrict__ ad_, unsigned int* __restrict__ menc, int E) {
  int e = blockIdx.x * blockDim.x + threadIdx.x;
  if (e >= E) return;
  int s = ei[e], d = ei[E + e];
#pragma unroll
  for (int h = 0; h < 4; ++h)
    atomicMax(&menc[d * 4 + h], fenc(lrelu(as_[s * 4 + h] + ad_[d * 4 + h])));
}

__global__ void att_self_exp_k(const float* __restrict__ as_, const float* __restrict__ ad_,
                               const unsigned int* __restrict__ menc, float* __restrict__ selfE,
                               float* __restrict__ denom, int N) {
  int idx = blockIdx.x * blockDim.x + threadIdx.x;
  if (idx >= N * 4) return;
  float ex = expf(lrelu(as_[idx] + ad_[idx]) - fdec(menc[idx]));
  selfE[idx] = ex; denom[idx] = ex;
}

__global__ void att_edge_exp_k(const int* __restrict__ ei, const float* __restrict__ as_,
                               const float* __restrict__ ad_, const unsigned int* __restrict__ menc,
                               float* __restrict__ eexp, float* __restrict__ denom, int E) {
  int e = blockIdx.x * blockDim.x + threadIdx.x;
  if (e >= E) return;
  int s = ei[e], d = ei[E + e];
#pragma unroll
  for (int h = 0; h < 4; ++h) {
    float ex = expf(lrelu(as_[s * 4 + h] + ad_[d * 4 + h]) - fdec(menc[d * 4 + h]));
    eexp[e * 4 + h] = ex;
    atomicAdd(&denom[d * 4 + h], ex);
  }
}

// init accumulators with self-loop contributions
__global__ void acc_init_k(const float* __restrict__ xw, const float* __restrict__ xh,
                           const float* __restrict__ dis, const float* __restrict__ selfE,
                           const float* __restrict__ denom, float* __restrict__ gcn,
                           float* __restrict__ gat, int N) {
  int idx = blockIdx.x * blockDim.x + threadIdx.x;
  if (idx >= N * D) return;
  int n = idx >> 8, c = idx & 255, h = c >> 6;
  float ns = dis[n] * dis[n];
  gcn[idx] = ns * xw[idx];
  gat[idx] = (selfE[n * 4 + h] / denom[n * 4 + h]) * xh[idx];
}

// one wave per edge: scatter-add gcn + gat messages (256 floats each)
__global__ void edge_scatter_k(const int* __restrict__ ei, const float* __restrict__ nrm,
                               const float* __restrict__ eexp, const float* __restrict__ denom,
                               const float* __restrict__ xw, const float* __restrict__ xh,
                               float* __restrict__ gcn, float* __restrict__ gat, int E) {
  int lane = threadIdx.x & 31;
  int e = blockIdx.x * (blockDim.x >> 5) + (threadIdx.x >> 5);
  if (e >= E) return;
  int s = ei[e], d = ei[E + e];
  float nm = nrm[e];
  float coef[4];
#pragma unroll
  for (int h = 0; h < 4; ++h) coef[h] = eexp[e * 4 + h] / denom[d * 4 + h];
  const float* xws = xw + (size_t)s * D;
  const float* xhs = xh + (size_t)s * D;
  float* gcd = gcn + (size_t)d * D;
  float* gtd = gat + (size_t)d * D;
#pragma unroll
  for (int i = 0; i < 8; ++i) {
    int c = lane + (i << 5);               // coalesced; head = c/64 = i/2
    atomicAdd(&gcd[c], nm * xws[c]);
    atomicAdd(&gtd[c], coef[i >> 1] * xhs[c]);
  }
}

// out = relu((gcn + gcn_b + gatproj) * bn_scale + bn_b); x += out
__global__ void combine_k(const float* __restrict__ gcn, const float* __restrict__ gcn_b,
                          const float* __restrict__ gp, const float* __restrict__ bn_g,
                          const float* __restrict__ bn_b, float* __restrict__ x, int N) {
  int idx = blockIdx.x * blockDim.x + threadIdx.x;
  if (idx >= N * D) return;
  int c = idx & 255;
  float o = gcn[idx] + gcn_b[c] + gp[idx];
  float scale = bn_g[c] * rsqrtf(1.f + 1e-5f);
  o = fmaxf(o * scale + bn_b[c], 0.f);
  x[idx] = o + x[idx];
}

// LayerNorm over D=256, one wave per node
__global__ void layernorm_k(const float* __restrict__ x, const float* __restrict__ g,
                            const float* __restrict__ b, float* __restrict__ out, int N) {
  int lane = threadIdx.x & 31;
  int n = blockIdx.x * (blockDim.x >> 5) + (threadIdx.x >> 5);
  if (n >= N) return;
  const float* row = x + (size_t)n * D;
  float vals[8], s = 0.f, s2 = 0.f;
#pragma unroll
  for (int i = 0; i < 8; ++i) {
    float v = row[lane + (i << 5)];
    vals[i] = v; s += v; s2 += v * v;
  }
#pragma unroll
  for (int off = 16; off >= 1; off >>= 1) {
    s  += __shfl_xor(s,  off, 32);
    s2 += __shfl_xor(s2, off, 32);
  }
  float mu = s * (1.f / D);
  float var = s2 * (1.f / D) - mu * mu;
  float inv = rsqrtf(var + 1e-5f);
#pragma unroll
  for (int i = 0; i < 8; ++i) {
    int c = lane + (i << 5);
    out[(size_t)n * D + c] = (vals[i] - mu) * inv * g[c] + b[c];
  }
}

__global__ void pool_init_k(float* cnt, float* sums, unsigned int* maxenc) {
  int idx = blockIdx.x * blockDim.x + threadIdx.x;
  if (idx < NGRAPH * D) { sums[idx] = 0.f; maxenc[idx] = fenc(-INFINITY); }
  if (idx < NGRAPH) cnt[idx] = 0.f;
}

__global__ void pool_cnt_k(const int* __restrict__ batch, float* cnt, int N) {
  int n = blockIdx.x * blockDim.x + threadIdx.x;
  if (n < N) atomicAdd(&cnt[batch[n]], 1.f);
}

__global__ void pool_acc_k(const float* __restrict__ node, const int* __restrict__ batch,
                           float* __restrict__ sums, unsigned int* __restrict__ maxenc, int N) {
  int idx = blockIdx.x * blockDim.x + threadIdx.x;
  if (idx >= N * D) return;
  int n = idx >> 8, c = idx & 255;
  int b = batch[n];
  float v = node[idx];
  atomicAdd(&sums[b * D + c], v);
  atomicMax(&maxenc[b * D + c], fenc(v));
}

__global__ void gf_build_k(const float* __restrict__ cnt, const float* __restrict__ sums,
                           const unsigned int* __restrict__ maxenc, float* __restrict__ gf) {
  int idx = blockIdx.x * blockDim.x + threadIdx.x;
  if (idx >= NGRAPH * 2 * D) return;
  int g = idx >> 9, c = idx & 511;
  if (c < D) {
    gf[idx] = sums[g * D + c] / fmaxf(cnt[g], 1.f);
  } else {
    float m = fdec(maxenc[g * D + (c - D)]);
    gf[idx] = (m > -3.0e38f) ? m : 0.f;    // empty group -> -inf -> 0
  }
}

// ---------------------------------------------------------------------------
static inline int cdiv(int a, int b) { return (a + b - 1) / b; }

static inline void launch_gemm(const float* A, const float* Abias, const float* W,
                               unsigned int* Bpk, const float* bias, float* C,
                               int M, int N, int K, bool relu, hipStream_t s) {
  int packN = (N / 16) * (K / 32) * 256;
  pack_B_k<<<cdiv(packN, 256), 256, 0, s>>>(W, Bpk, N, K);
  int tasks = (M / 16) * (N / 64);
  int blocks = cdiv(tasks, 8);
  if (Abias) {
    if (relu) gemm_wmma_bf16<1, 1><<<blocks, 256, 0, s>>>(A, Abias, Bpk, bias, C, M, N, K);
    else      gemm_wmma_bf16<0, 1><<<blocks, 256, 0, s>>>(A, Abias, Bpk, bias, C, M, N, K);
  } else {
    if (relu) gemm_wmma_bf16<1, 0><<<blocks, 256, 0, s>>>(A, Abias, Bpk, bias, C, M, N, K);
    else      gemm_wmma_bf16<0, 0><<<blocks, 256, 0, s>>>(A, Abias, Bpk, bias, C, M, N, K);
  }
}

extern "C" void kernel_launch(void* const* d_in, const int* in_sizes, int n_in,
                              void* d_out, int out_size, void* d_ws, size_t ws_size,
                              hipStream_t stream) {
  // setup_inputs() dict order, params flattened depth-first in insertion order:
  // 0 func_embeddings, 1 edge_emb, 2 in_W, 3 in_b, 4 ew_W1, 5 ew_b1, 6 ew_W2,
  // 7 ew_b2, 8 ln_g, 9 ln_b, 10 ro_W1, 11 ro_b1, 12 ro_W2, 13 ro_b2,
  // 14+10l: gcn_W,gcn_b,gat_W,att_src,att_dst,gat_b,proj_W,proj_b,bn_g,bn_b (l=0..2)
  // 44 edge_index, 45 edge_attr, 46 batch
  const float* F        = (const float*)d_in[0];
  const float* edge_emb = (const float*)d_in[1];
  const float* in_W     = (const float*)d_in[2];
  const float* in_b     = (const float*)d_in[3];
  const float* ew_W1    = (const float*)d_in[4];
  const float* ew_b1    = (const float*)d_in[5];
  const float* ew_W2    = (const float*)d_in[6];
  const float* ew_b2    = (const float*)d_in[7];
  const float* ln_g     = (const float*)d_in[8];
  const float* ln_b     = (const float*)d_in[9];
  const float* ro_W1    = (const float*)d_in[10];
  const float* ro_b1    = (const float*)d_in[11];
  const float* ro_W2    = (const float*)d_in[12];
  const float* ro_b2    = (const float*)d_in[13];
  const int*   eidx     = (const int*)d_in[44];
  const int*   eattr    = (const int*)d_in[45];
  const int*   batch    = (const int*)d_in[46];

  const int N = in_sizes[46];   // 100000
  const int E = in_sizes[45];   // 300000
  const size_t ND = (size_t)N * D;

  // workspace carve-out
  char* p = (char*)d_ws;
  auto carve = [&](size_t bytes) { void* r = p; p += (bytes + 255) & ~(size_t)255; return r; };
  float* x     = (float*)carve(ND * 4);
  float* xw    = (float*)carve(ND * 4);   // reused for gatproj
  float* xh    = (float*)carve(ND * 4);
  float* gcn   = (float*)carve(ND * 4);
  float* gat   = (float*)carve(ND * 4);
  float* ew    = (float*)carve((size_t)E * 4);
  float* deg   = (float*)carve((size_t)N * 4);      // becomes D^{-1/2} in place
  float* nrm   = (float*)carve((size_t)E * 4);
  float* as_   = (float*)carve((size_t)N * 4 * 4);
  float* ad_   = (float*)carve((size_t)N * 4 * 4);
  unsigned int* menc = (unsigned int*)carve((size_t)N * 4 * 4);
  float* selfE = (float*)carve((size_t)N * 4 * 4);
  float* denom = (float*)carve((size_t)N * 4 * 4);
  float* eexp  = (float*)carve((size_t)E * 4 * 4);
  float* cnt   = (float*)carve(NGRAPH * 4);
  float* sums  = (float*)carve(NGRAPH * D * 4);
  unsigned int* maxenc = (unsigned int*)carve(NGRAPH * D * 4);
  float* gf    = (float*)carve(NGRAPH * 2 * D * 4);
  float* g1    = (float*)carve(NGRAPH * D * 4);
  unsigned int* Bpk = (unsigned int*)carve((size_t)(2 * D) * D * 2);  // max K=512: K*N/2 uints

  float* node_out  = (float*)d_out;                 // [N, 256]
  float* graph_out = (float*)d_out + ND;            // [64, 256]

  // ---- edge weights + GCN normalization (shared across layers) ----
  deg_init_k<<<cdiv(N, 256), 256, 0, stream>>>(deg, N);
  edge_weight_k<<<cdiv(E, 128), 128, 0, stream>>>(eattr, eidx, edge_emb,
                                                  ew_W1, ew_b1, ew_W2, ew_b2, ew, deg, E);
  dis_k<<<cdiv(N, 256), 256, 0, stream>>>(deg, N);
  norm_k<<<cdiv(E, 256), 256, 0, stream>>>(eidx, ew, deg, nrm, E);

  // ---- input projection: x = F @ in_W + in_b ----
  launch_gemm(F, nullptr, in_W, Bpk, in_b, x, N, D, D, false, stream);

  // ---- 3 GNN layers ----
  for (int l = 0; l < 3; ++l) {
    const int base = 14 + 10 * l;
    const float* gcn_W  = (const float*)d_in[base + 0];
    const float* gcn_b  = (const float*)d_in[base + 1];
    const float* gat_W  = (const float*)d_in[base + 2];
    const float* att_s  = (const float*)d_in[base + 3];
    const float* att_d  = (const float*)d_in[base + 4];
    const float* gat_b  = (const float*)d_in[base + 5];
    const float* proj_W = (const float*)d_in[base + 6];
    const float* proj_b = (const float*)d_in[base + 7];
    const float* bn_g   = (const float*)d_in[base + 8];
    const float* bn_b   = (const float*)d_in[base + 9];

    launch_gemm(x, nullptr, gcn_W, Bpk, nullptr, xw, N, D, D, false, stream);
    launch_gemm(x, nullptr, gat_W, Bpk, nullptr, xh, N, D, D, false, stream);

    att_logits_k<<<cdiv(N * 4, 256), 256, 0, stream>>>(xh, att_s, att_d, as_, ad_, N);
    att_minit_k<<<cdiv(N * 4, 256), 256, 0, stream>>>(as_, ad_, menc, N);
    att_edge_max_k<<<cdiv(E, 256), 256, 0, stream>>>(eidx, as_, ad_, menc, E);
    att_self_exp_k<<<cdiv(N * 4, 256), 256, 0, stream>>>(as_, ad_, menc, selfE, denom, N);
    att_edge_exp_k<<<cdiv(E, 256), 256, 0, stream>>>(eidx, as_, ad_, menc, eexp, denom, E);

    acc_init_k<<<cdiv(N * D, 256), 256, 0, stream>>>(xw, xh, deg, selfE, denom, gcn, gat, N);
    edge_scatter_k<<<cdiv(E, 8), 256, 0, stream>>>(eidx, nrm, eexp, denom, xw, xh, gcn, gat, E);

    // gatproj = (gat + gat_b) @ proj_W + proj_b  (reuse xw buffer)
    launch_gemm(gat, gat_b, proj_W, Bpk, proj_b, xw, N, D, D, false, stream);
    combine_k<<<cdiv(N * D, 256), 256, 0, stream>>>(gcn, gcn_b, xw, bn_g, bn_b, x, N);
  }

  // ---- LayerNorm -> node output ----
  layernorm_k<<<cdiv(N, 8), 256, 0, stream>>>(x, ln_g, ln_b, node_out, N);

  // ---- global mean/max pooling + readout MLP ----
  pool_init_k<<<cdiv(NGRAPH * D, 256), 256, 0, stream>>>(cnt, sums, maxenc);
  pool_cnt_k<<<cdiv(N, 256), 256, 0, stream>>>(batch, cnt, N);
  pool_acc_k<<<cdiv(N * D, 256), 256, 0, stream>>>(node_out, batch, sums, maxenc, N);
  gf_build_k<<<cdiv(NGRAPH * 2 * D, 256), 256, 0, stream>>>(cnt, sums, maxenc, gf);

  launch_gemm(gf, nullptr, ro_W1, Bpk, ro_b1, g1, NGRAPH, D, 2 * D, true, stream);
  launch_gemm(g1, nullptr, ro_W2, Bpk, ro_b2, graph_out, NGRAPH, D, D, false, stream);

  (void)n_in; (void)out_size; (void)ws_size;
}
